// Encoder_53077205844214
// MI455X (gfx1250) — matrix operand order repaired and checked by reading, not compiled, not run on hardware
//
#include <hip/hip_runtime.h>
#include <hip/hip_bf16.h>
#include <math.h>

typedef __bf16 bf16_t;
typedef __attribute__((ext_vector_type(16))) __bf16        v16bf;
typedef __attribute__((ext_vector_type(8)))  float         v8f;
typedef __attribute__((ext_vector_type(4)))  unsigned int  u32x4;
typedef __attribute__((ext_vector_type(4)))  float         f32x4;

#define BM  128
#define BN  128
#define BK  32
#define LDT 48   // LDS row stride in bf16 elems: 96 bytes, 16B-aligned, bank-spread

// ternary pair -> packed bf16 bit patterns {0, +1, -1} = {0x0000, 0x3F80, 0xBF80}
__device__ __forceinline__ unsigned int tern_pair(float w0, float n0,
                                                  float w1, float n1) {
    float q0 = w0 - n0, q1 = w1 - n1;
    unsigned int lo = (q0 > 1.0f) ? 0x3F80u : ((q0 < -1.0f) ? 0xBF80u : 0u);
    unsigned int hi = (q1 > 1.0f) ? 0x3F80u : ((q1 < -1.0f) ? 0xBF80u : 0u);
    return lo | (hi << 16);
}

// branch-free tanh: prefer HW V_TANH_F32, else 1 - 2/(exp2(2*log2e*x)+1)
__device__ __forceinline__ float fast_tanh(float x) {
#if __has_builtin(__builtin_amdgcn_tanhf)
    return __builtin_amdgcn_tanhf(x);
#elif __has_builtin(__builtin_amdgcn_exp2f) && __has_builtin(__builtin_amdgcn_rcpf)
    float e = __builtin_amdgcn_exp2f(x * 2.8853900817779268f); // 2*log2(e)
    return 1.0f - 2.0f * __builtin_amdgcn_rcpf(e + 1.0f);
#else
    float e = exp2f(x * 2.8853900817779268f);
    return 1.0f - 2.0f / (e + 1.0f);
#endif
}

// ---- pipeline stage helpers -------------------------------------------------

// fp32 A tile: global -> regs
__device__ __forceinline__ void a_g2r_f32(const float* A, size_t rowbase, int kcol,
                                          f32x4 fa[4]) {
    const f32x4* s = (const f32x4*)(A + rowbase + kcol);
    fa[0] = s[0]; fa[1] = s[1]; fa[2] = s[2]; fa[3] = s[3];
}

// fp32 A tile: regs -> LDS (converted to bf16)
__device__ __forceinline__ void a_r2s_f32(bf16_t* ldsA, int arow, int ahalf,
                                          const f32x4 fa[4]) {
    union { bf16_t h[16]; u32x4 u[2]; } pk;
    #pragma unroll
    for (int j = 0; j < 4; ++j)
        #pragma unroll
        for (int i = 0; i < 4; ++i) pk.h[j * 4 + i] = (bf16_t)fa[j][i];
    u32x4* dst = (u32x4*)(ldsA + arow * LDT + ahalf);
    dst[0] = pk.u[0]; dst[1] = pk.u[1];
}

// bf16 A tile: async global -> LDS copy (32B per lane), tracked by ASYNCcnt.
// Low 32 bits of a generic __shared__ pointer are the LDS byte offset
// (shared aperture lives entirely in the high word of the flat address).
__device__ __forceinline__ void a_g2s_async(const bf16_t* gsrc, bf16_t* ldst) {
    unsigned int l0 = (unsigned int)(unsigned long long)(uintptr_t)(void*)ldst;
    unsigned long long g = (unsigned long long)(uintptr_t)gsrc;
    asm volatile("global_load_async_to_lds_b128 %0, %1, off\n\t"
                 "global_load_async_to_lds_b128 %0, %1, off offset:16"
                 :: "v"(l0), "v"(g) : "memory");
}

__device__ __forceinline__ void async_wait0() {
    asm volatile("s_wait_asynccnt 0" ::: "memory");
}

// B tile: global -> regs (2 k-rows x 8 n each, weights + noise)
__device__ __forceinline__ void b_g2r(const float* W, const float* Noise,
                                      size_t r0, int N, f32x4 bw[4], f32x4 bz[4]) {
    const f32x4* w0 = (const f32x4*)(W + r0);
    const f32x4* w1 = (const f32x4*)(W + r0 + N);
    const f32x4* z0 = (const f32x4*)(Noise + r0);
    const f32x4* z1 = (const f32x4*)(Noise + r0 + N);
    bw[0] = w0[0]; bw[1] = w0[1]; bw[2] = w1[0]; bw[3] = w1[1];
    bz[0] = z0[0]; bz[1] = z0[1]; bz[2] = z1[0]; bz[3] = z1[1];
}

// B tile: ternarize + transpose into LDS [n][k], k-pairs packed per dword
__device__ __forceinline__ void b_r2s(bf16_t* ldsB, int bng, int bkp,
                                      const f32x4 bw[4], const f32x4 bz[4]) {
    #pragma unroll
    for (int j = 0; j < 8; ++j) {
        unsigned int u = tern_pair(bw[j >> 2][j & 3],       bz[j >> 2][j & 3],
                                   bw[2 + (j >> 2)][j & 3], bz[2 + (j >> 2)][j & 3]);
        *(unsigned int*)(ldsB + (bng + j) * LDT + 2 * bkp) = u;
    }
}

// D = A(MxK) @ tern(W - Noise)(KxN); epilogue: out = f(acc*scale[n] + bias[n])
template<bool A_BF16, bool TANH_OUT, typename OutT>
__global__ __launch_bounds__(256)
void ternary_gemm_kernel(const void* __restrict__ Ain,
                         const float* __restrict__ W,
                         const float* __restrict__ Noise,
                         const float* __restrict__ scale,
                         const float* __restrict__ bias,
                         OutT* __restrict__ Out,
                         int M, int N, int K)
{
    __shared__ bf16_t ldsA[2][BM * LDT];
    __shared__ bf16_t ldsB[2][BN * LDT];

    const int tid  = threadIdx.x;
    const int lane = tid & 31;
    const int wv   = tid >> 5;           // 8 waves
    const int wm   = (wv >> 1) * 32;     // wave M offset (4 waves in M)
    const int wn   = (wv & 1)  * 64;     // wave N offset (2 waves in N)
    const int m0   = blockIdx.y * BM;
    const int n0   = blockIdx.x * BN;

    // A stager coords: 2 threads per row, 16 k-elems each
    const int arow  = tid >> 1;
    const int ahalf = (tid & 1) * 16;
    // B stager coords: thread owns k-rows {2*bkp, 2*bkp+1} x 8 n-cols
    const int bkp = tid >> 4;            // 0..15
    const int bng = (tid & 15) * 8;      // 0..120

    const size_t arowbase = (size_t)(m0 + arow) * K;
    const size_t bcolbase = (size_t)n0 + bng;
    const bf16_t* Abf = (const bf16_t*)Ain;
    const float*  Afp = (const float*)Ain;

    v8f acc[2][4] = {};

    f32x4 fa[4];
    f32x4 bw[4], bz[4];

    const int NT = K / BK;

    // ---- prologue: stage tile 0 ----
    if (A_BF16) {
        a_g2s_async(Abf + arowbase + ahalf, &ldsA[0][arow * LDT + ahalf]);
    } else {
        a_g2r_f32(Afp, arowbase, ahalf, fa);
    }
    b_g2r(W, Noise, (size_t)(2 * bkp) * N + bcolbase, N, bw, bz);
    if (!A_BF16) a_r2s_f32(ldsA[0], arow, ahalf, fa);
    b_r2s(ldsB[0], bng, bkp, bw, bz);
    if (A_BF16) async_wait0();
    __syncthreads();

    const int fr  = lane & 15;
    const int fka = (lane >> 4) * 16;   // A: K 0-15 / 16-31 contiguous
    const int fkb = (lane >> 4) * 16;   // B: K 0-15 / 16-31 contiguous

    for (int kt = 0; kt < NT; ++kt) {
        const int cur = kt & 1;
        const int nxt = cur ^ 1;
        const bool more = (kt + 1) < NT;

        // issue next tile's global traffic before the matrix ops (latency overlap)
        if (more) {
            const int k1 = (kt + 1) * BK;
            if (A_BF16) {
                a_g2s_async(Abf + arowbase + k1 + ahalf,
                            &ldsA[nxt][arow * LDT + ahalf]);
            } else {
                a_g2r_f32(Afp, arowbase, k1 + ahalf, fa);
            }
            b_g2r(W, Noise, (size_t)(k1 + 2 * bkp) * N + bcolbase, N, bw, bz);
        }
        if (kt + 2 < NT) {   // prefetch weight stream two tiles ahead -> global_prefetch_b8
            const size_t r2 = (size_t)((kt + 2) * BK + 2 * bkp) * N + bcolbase;
            __builtin_prefetch(W + r2, 0, 0);
            __builtin_prefetch(Noise + r2, 0, 0);
        }

        // ---- fragments in contiguous k order (k = 16*half + s, A and B alike) & 8 WMMAs on current buffer ----
        const bf16_t* lA = ldsA[cur];
        const bf16_t* lB = ldsB[cur];
        v16bf afrag[2], bfrag[4];
        #pragma unroll
        for (int i = 0; i < 2; ++i) {
            union { u32x4 u[2]; v16bf v; } f;
            const bf16_t* p = lA + (wm + i * 16 + fr) * LDT;
            f.u[0] = *(const u32x4*)(p + fka);
            f.u[1] = *(const u32x4*)(p + fka + 8);
            afrag[i] = f.v;
        }
        #pragma unroll
        for (int j = 0; j < 4; ++j) {
            union { u32x4 u[2]; v16bf v; } f;
            const bf16_t* p = lB + (wn + j * 16 + fr) * LDT + fkb;
            f.u[0] = ((const u32x4*)p)[0];
            f.u[1] = ((const u32x4*)p)[1];
            bfrag[j] = f.v;
        }
        #pragma unroll
        for (int i = 0; i < 2; ++i)
            #pragma unroll
            for (int j = 0; j < 4; ++j)
                acc[i][j] = __builtin_amdgcn_wmma_f32_16x16x32_bf16(
                    false, afrag[i], false, bfrag[j],
                    (short)0, acc[i][j], false, false);

        // ---- convert/ternarize + stage next tile into other buffer ----
        if (more) {
            if (!A_BF16) a_r2s_f32(ldsA[nxt], arow, ahalf, fa);
            b_r2s(ldsB[nxt], bng, bkp, bw, bz);
            if (A_BF16) async_wait0();
        }
        __syncthreads();
    }

    // ---- epilogue: C/D layout = lane(0-15)->N col, VGPR v -> M row v + 8*laneHi ----
    const int col   = lane & 15;
    const int rbase = (lane >> 4) * 8;
    #pragma unroll
    for (int j = 0; j < 4; ++j) {
        const int gn = n0 + wn + j * 16 + col;
        const float s  = scale[gn];
        const float bb = bias[gn];
        #pragma unroll
        for (int i = 0; i < 2; ++i) {
            #pragma unroll
            for (int v = 0; v < 8; ++v) {
                const int gm = m0 + wm + i * 16 + rbase + v;
                float r = acc[i][j][v] * s + bb;
                if (TANH_OUT) r = fast_tanh(r);
                Out[(size_t)gm * N + gn] = (OutT)r;
            }
        }
    }
}

extern "C" void kernel_launch(void* const* d_in, const int* in_sizes, int n_in,
                              void* d_out, int out_size, void* d_ws, size_t ws_size,
                              hipStream_t stream) {
    const float* x      = (const float*)d_in[0];
    const float* w1     = (const float*)d_in[1];
    const float* s1     = (const float*)d_in[2];
    const float* b1     = (const float*)d_in[3];
    const float* w2     = (const float*)d_in[4];
    const float* s2     = (const float*)d_in[5];
    const float* b2     = (const float*)d_in[6];
    const float* noise1 = (const float*)d_in[7];
    const float* noise2 = (const float*)d_in[8];

    const int B = 2048, DIN = 3072, DHID = 16384, DOUT = 1024;

    // intermediate h kept as bf16 in workspace: 2048*16384*2 = 64 MB
    bf16_t* h = (bf16_t*)d_ws;

    dim3 blk(256);
    // layer 1: h = tanh(x @ tern(w1-n1) * s1 + b1)   -> bf16
    ternary_gemm_kernel<false, true, bf16_t>
        <<<dim3(DHID / BN, B / BM), blk, 0, stream>>>(
            x, w1, noise1, s1, b1, h, B, DHID, DIN);
    // layer 2: out = h @ tern(w2-n2) * s2 + b2       -> f32
    ternary_gemm_kernel<true, false, float>
        <<<dim3(DOUT / BN, B / BM), blk, 0, stream>>>(
            h, w2, noise2, s2, b2, (float*)d_out, B, DOUT, DHID);
}
